// self_LSTM_sparse_attn_predict_28879360098671
// MI455X (gfx1250) — matrix operand order repaired and checked by reading, not compiled, not run on hardware
//
#include <hip/hip_runtime.h>
#include <hip/hip_bf16.h>
#include <stdint.h>
#include <stddef.h>

#define B_    64
#define T_    512
#define I_    128
#define H_    256
#define G4_   1024   // 4*H
#define C_    128
#define TOPK_ 10
#define EPSF  1e-7f
#define MSLOT 513    // T+1

typedef _Float16 h8v  __attribute__((ext_vector_type(8)));
typedef _Float16 h16v __attribute__((ext_vector_type(16)));
typedef float    f8v  __attribute__((ext_vector_type(8)));

__device__ __forceinline__ h16v cat16(h8v lo, h8v hi) {
  return __builtin_shufflevector(lo, hi, 0,1,2,3,4,5,6,7,8,9,10,11,12,13,14,15);
}
__device__ __forceinline__ float sigm(float x) { return 1.0f / (1.0f + __expf(-x)); }

__device__ __forceinline__ float wred_sum(float v) {
  #pragma unroll
  for (int o = 16; o > 0; o >>= 1) v += __shfl_xor(v, o, 32);
  return v;
}
__device__ __forceinline__ float wred_max(float v) {
  #pragma unroll
  for (int o = 16; o > 0; o >>= 1) v = fmaxf(v, __shfl_xor(v, o, 32));
  return v;
}
__device__ __forceinline__ int wred_sumi(int v) {
  #pragma unroll
  for (int o = 16; o > 0; o >>= 1) v += __shfl_xor(v, o, 32);
  return v;
}

// ---------------- utility kernels ----------------

__global__ void k_zero(float* p, size_t n) {
  for (size_t i = (size_t)blockIdx.x * blockDim.x + threadIdx.x; i < n;
       i += (size_t)gridDim.x * blockDim.x)
    p[i] = 0.0f;
}

__global__ void k_addbias(const float* __restrict__ a, const float* __restrict__ b,
                          float* __restrict__ o) {
  int i = blockIdx.x * blockDim.x + threadIdx.x;
  if (i < G4_) o[i] = a[i] + b[i];
}

// x (B,T,I) f32  ->  x16 [t][b][k] f16   (row r = t*B+b matches gx row order)
__global__ void k_cvtx(const float* __restrict__ x, _Float16* __restrict__ x16) {
  int id = blockIdx.x * blockDim.x + threadIdx.x;   // == output index
  int k  = id & (I_ - 1);
  int tb = id >> 7;           // t*64 + b
  int b  = tb & (B_ - 1);
  int t  = tb >> 6;
  x16[id] = (_Float16)x[((size_t)b * T_ + t) * I_ + k];
}

// Pack W (N x K, row-major f32) into WMMA-B lane layout:
// tile (kt,nt): lane = l, n = nt*16+(l&15), K = kt*32 + ((l>=16)?8:0) + (j<8 ? j : j+8), j=0..15
__global__ void k_packb(const float* __restrict__ W, _Float16* __restrict__ out,
                        int N, int K) {
  int id = blockIdx.x * blockDim.x + threadIdx.x;
  int ntiles = N >> 4;
  int total  = (K >> 5) * ntiles * 32;
  if (id >= total) return;
  int lane = id & 31;
  int tile = id >> 5;
  int kt = tile / ntiles;
  int nt = tile - kt * ntiles;
  int n  = nt * 16 + (lane & 15);
  int kb = kt * 32 + ((lane >> 4) << 3);
  _Float16* o = out + (size_t)id * 16;
  const float* w = W + (size_t)n * K + kb;
  #pragma unroll
  for (int j = 0; j < 16; ++j) o[j] = (_Float16)w[j < 8 ? j : j + 8];
}

// ---------------- generic WMMA GEMM:  C = A(f16,MxK) * Bpacked^ + bias ----------------
__global__ void k_gemm16(const _Float16* __restrict__ A, const _Float16* __restrict__ Bp,
                         const float* __restrict__ bias, float* __restrict__ Cm,
                         int M, int N, int K) {
  const int tiles_n = N >> 4;
  const int total   = (M >> 4) * tiles_n;
  int tile = blockIdx.x * (blockDim.x >> 5) + (threadIdx.x >> 5);
  if (tile >= total) return;                 // wave-uniform: EXEC stays all-ones
  const int lane = threadIdx.x & 31;
  const int mt = tile / tiles_n;
  const int nt = tile - mt * tiles_n;
  const int l15  = lane & 15;
  const int m    = mt * 16 + l15;
  const int koff = (lane >> 4) << 3;         // A layout: 0 / 8
  const int n    = nt * 16 + l15;
  float bv = bias ? bias[n] : 0.0f;
  f8v acc;
  #pragma unroll
  for (int r = 0; r < 8; ++r) acc[r] = bv;
  const int ktiles = K >> 5;
  const _Float16* arow = A + (size_t)m * K;
  for (int kt = 0; kt < ktiles; ++kt) {
    h8v alo = *(const h8v*)(arow + kt * 32 + koff);
    h8v ahi = *(const h8v*)(arow + kt * 32 + koff + 16);
    h16v a  = cat16(alo, ahi);
    h16v b  = *(const h16v*)(Bp + (((size_t)kt * tiles_n + nt) * 32 + lane) * 16);
    acc = __builtin_amdgcn_wmma_f32_16x16x32_f16(false, a, false, b, (short)0, acc,
                                                 false, false);
  }
  const int rowb = mt * 16 + ((lane >> 4) << 3);
  #pragma unroll
  for (int r = 0; r < 8; ++r) Cm[(size_t)(rowb + r) * N + n] = acc[r];
}

// ---------------- per-step kernel 1: gates GEMM (h @ W_hh^T) + LSTM pointwise ----------------
// grid 64 blocks x 128 threads. block = (mt in 0..3, ht in 0..15); wave g computes gate g tile.
__global__ void k_lstm_step(const _Float16* __restrict__ h16,
                            const _Float16* __restrict__ Whh_p,
                            const float* __restrict__ gx,    // [T][B][4H] (x-part + biases)
                            float* __restrict__ cbuf,        // [B][H]
                            float* __restrict__ hl,          // [B][H] h_lstm f32
                            _Float16* __restrict__ hlstm16,  // [B][T][H] archive for pred GEMM
                            int t) {
  __shared__ float g_lds[4][16][16];
  const int mt   = blockIdx.x >> 4;
  const int ht   = blockIdx.x & 15;
  const int wave = threadIdx.x >> 5;        // gate id
  const int lane = threadIdx.x & 31;
  const int m    = mt * 16 + (lane & 15);
  const int koff = (lane >> 4) << 3;
  const int nt   = wave * 16 + ht;          // tile in N=1024 (64 tiles)
  f8v acc;
  #pragma unroll
  for (int r = 0; r < 8; ++r) acc[r] = 0.0f;
  const _Float16* arow = h16 + (size_t)m * H_;
  #pragma unroll
  for (int kt = 0; kt < 8; ++kt) {          // K = 256
    h8v alo = *(const h8v*)(arow + kt * 32 + koff);
    h8v ahi = *(const h8v*)(arow + kt * 32 + koff + 16);
    h16v a  = cat16(alo, ahi);
    h16v b  = *(const h16v*)(Whh_p + (((size_t)kt * 64 + nt) * 32 + lane) * 16);
    acc = __builtin_amdgcn_wmma_f32_16x16x32_f16(false, a, false, b, (short)0, acc,
                                                 false, false);
  }
  const int col = lane & 15, rb = (lane >> 4) << 3;
  #pragma unroll
  for (int r = 0; r < 8; ++r) g_lds[wave][rb + r][col] = acc[r];
  __syncthreads();
  // pointwise LSTM over the 16x16 (b,h) tile; 128 threads -> 2 elements each
  for (int e = threadIdx.x; e < 256; e += 128) {
    int mr = e >> 4, cc = e & 15;
    int b  = mt * 16 + mr;
    int hh = ht * 16 + cc;
    const float* gxr = gx + ((size_t)t * B_ + b) * G4_;
    float ig = g_lds[0][mr][cc] + gxr[0 * H_ + hh];
    float fg = g_lds[1][mr][cc] + gxr[1 * H_ + hh];
    float gg = g_lds[2][mr][cc] + gxr[2 * H_ + hh];
    float og = g_lds[3][mr][cc] + gxr[3 * H_ + hh];
    float co = cbuf[b * H_ + hh];
    float cn = sigm(fg) * co + sigm(ig) * tanhf(gg);
    float hv = sigm(og) * tanhf(cn);
    cbuf[b * H_ + hh] = cn;
    hl[b * H_ + hh]   = hv;
    hlstm16[((size_t)b * T_ + t) * H_ + hh] = (_Float16)hv;
  }
}

// ---------------- per-step kernel 2: scores + top-k threshold + attn weights ----------------
// grid 64 blocks x 32 threads (one wave per batch row)
__global__ void k_attn_score(const float* __restrict__ hl, const float* __restrict__ s2,
                             const float* __restrict__ w_t, float* __restrict__ attnw,
                             int t) {
  const int b = blockIdx.x, lane = threadIdx.x;
  float p = 0.0f;
  #pragma unroll
  for (int j = 0; j < 8; ++j) {
    int h = lane + 32 * j;
    p += tanhf(hl[b * H_ + h]) * w_t[h];    // w1
  }
  float sc1 = wred_sum(p);
  float sc[16];
  #pragma unroll
  for (int j = 0; j < 16; ++j) {
    int m = lane + 32 * j;                  // covers 0..511
    sc[j] = (m <= t) ? (sc1 + s2[b * MSLOT + m]) : -1e9f;
  }
  if (t + 1 <= TOPK_) {                     // raw = scores * mask
    #pragma unroll
    for (int j = 0; j < 16; ++j) {
      int m = lane + 32 * j;
      if (m <= t) attnw[b * MSLOT + m] = sc[j];
    }
  } else {
    // find 10th-largest (duplicate-aware) via iterative max-remove
    float tmp[16];
    #pragma unroll
    for (int j = 0; j < 16; ++j) tmp[j] = sc[j];
    int removed = 0;
    float delta = -3e9f;
    for (int it = 0; it < TOPK_; ++it) {
      float lm = -3e9f;
      #pragma unroll
      for (int j = 0; j < 16; ++j) lm = fmaxf(lm, tmp[j]);
      float Mx = wred_max(lm);
      int lc = 0;
      #pragma unroll
      for (int j = 0; j < 16; ++j) lc += (tmp[j] == Mx) ? 1 : 0;
      int cnt = wred_sumi(lc);
      if (removed + cnt >= TOPK_) { delta = Mx + EPSF; break; }
      removed += cnt;
      #pragma unroll
      for (int j = 0; j < 16; ++j) if (tmp[j] == Mx) tmp[j] = -3e9f;
    }
    float sp[16];
    float s = 0.0f;
    #pragma unroll
    for (int j = 0; j < 16; ++j) { sp[j] = fmaxf(sc[j] - delta, 0.0f); s += sp[j]; }
    float inv = 1.0f / (wred_sum(s) + EPSF);
    #pragma unroll
    for (int j = 0; j < 16; ++j) {
      int m = lane + 32 * j;
      if (m <= t) attnw[b * MSLOT + m] = sp[j] * inv;
    }
  }
}

// ---------------- per-step kernel 3: attn_c, h_new, mem/s2 update, feat staging, viz ----------------
// grid 67 blocks x 256 threads: blocks 0..63 per batch row; blocks 64..66 compute viz row
__global__ void k_attn_apply(const float* __restrict__ attnw, const float* __restrict__ hl,
                             float* __restrict__ memb, const float* __restrict__ w_t,
                             _Float16* __restrict__ h16b, _Float16* __restrict__ feat16,
                             float* __restrict__ houts, float* __restrict__ viz,
                             float* __restrict__ s2, int t) {
  if (blockIdx.x < 64) {
    const int b = blockIdx.x, tid = threadIdx.x;     // tid = h index
    __shared__ float sa[512];
    __shared__ float red[256];
    for (int m = tid; m <= t; m += 256) sa[m] = attnw[b * MSLOT + m];
    __syncthreads();
    const float* mb = memb + (size_t)b * MSLOT * H_;
    float acc = 0.0f;
    for (int m = 0; m <= t; ++m) acc += sa[m] * mb[(size_t)m * H_ + tid];
    float hn = hl[b * H_ + tid] + acc;
    memb[(size_t)b * MSLOT * H_ + (size_t)(t + 1) * H_ + tid] = hn;
    h16b[b * H_ + tid] = (_Float16)hn;
    size_t ro = (size_t)b * T_ + t;
    houts[ro * H_ + tid]            = hn;
    feat16[ro * 2 * H_ + tid]       = (_Float16)hn;
    feat16[ro * 2 * H_ + H_ + tid]  = (_Float16)acc;
    red[tid] = tanhf(hn) * w_t[H_ + tid];            // w2
    __syncthreads();
    for (int o = 128; o > 0; o >>= 1) {
      if (tid < o) red[tid] += red[tid + o];
      __syncthreads();
    }
    if (tid == 0) s2[b * MSLOT + (t + 1)] = red[0];
  } else {
    int m = (blockIdx.x - 64) * 256 + threadIdx.x;
    if (m < MSLOT) {
      float s = 0.0f;
      for (int b = 0; b < B_; ++b) s += attnw[b * MSLOT + m];
      viz[(size_t)t * MSLOT + m] = s * (1.0f / 64.0f);
    }
  }
}

// ---------------- host launch ----------------
extern "C" void kernel_launch(void* const* d_in, const int* in_sizes, int n_in,
                              void* d_out, int out_size, void* d_ws, size_t ws_size,
                              hipStream_t stream) {
  (void)in_sizes; (void)n_in; (void)out_size; (void)ws_size;
  const float* x    = (const float*)d_in[0];
  const float* Wih  = (const float*)d_in[1];
  const float* Whh  = (const float*)d_in[2];
  const float* bih  = (const float*)d_in[3];
  const float* bhh  = (const float*)d_in[4];
  const float* w_t  = (const float*)d_in[5];
  const float* Wp   = (const float*)d_in[6];
  const float* bp   = (const float*)d_in[7];
  const float* Wfc  = (const float*)d_in[8];
  const float* bfc  = (const float*)d_in[9];

  // output regions (tuple order: out, viz, predicted_all, h_outs)
  float* out_fc = (float*)d_out;
  float* viz    = out_fc + (size_t)B_ * T_ * C_;          // 4,194,304
  float* pred   = viz + (size_t)T_ * MSLOT;               // +262,656
  float* houts  = pred + (size_t)B_ * T_ * H_;            // +8,388,608

  // workspace carve (256B aligned)
  size_t off = 0;
  char* base = (char*)d_ws;
  auto carve = [&](size_t bytes) -> char* {
    char* p = base + off;
    off += (bytes + 255) & ~(size_t)255;
    return p;
  };
  float*    gx      = (float*)carve((size_t)T_ * B_ * G4_ * 4);      // 134 MB
  float*    memb    = (float*)carve((size_t)B_ * MSLOT * H_ * 4);    // 33.6 MB
  float*    s2      = (float*)carve((size_t)B_ * MSLOT * 4);
  float*    attnw   = (float*)carve((size_t)B_ * MSLOT * 4);
  float*    cbuf    = (float*)carve((size_t)B_ * H_ * 4);
  float*    hl      = (float*)carve((size_t)B_ * H_ * 4);
  float*    bsum    = (float*)carve((size_t)G4_ * 4);
  _Float16* h16b    = (_Float16*)carve((size_t)B_ * H_ * 2);
  _Float16* x16     = (_Float16*)carve((size_t)B_ * T_ * I_ * 2);
  _Float16* hlstm16 = (_Float16*)carve((size_t)B_ * T_ * H_ * 2);
  _Float16* feat16  = (_Float16*)carve((size_t)B_ * T_ * 2 * H_ * 2);
  _Float16* Wih_p   = (_Float16*)carve((size_t)(I_ / 32) * (G4_ / 16) * 512 * 2);
  _Float16* Whh_p   = (_Float16*)carve((size_t)(H_ / 32) * (G4_ / 16) * 512 * 2);
  _Float16* Wp_p    = (_Float16*)carve((size_t)(H_ / 32) * (H_ / 16) * 512 * 2);
  _Float16* Wfc_p   = (_Float16*)carve((size_t)(2 * H_ / 32) * (C_ / 16) * 512 * 2);

  // ---- state init (every call: deterministic) ----
  k_zero<<<1024, 256, 0, stream>>>(memb, (size_t)B_ * MSLOT * H_);
  k_zero<<<64,   256, 0, stream>>>(s2,    (size_t)B_ * MSLOT);
  k_zero<<<64,   256, 0, stream>>>(attnw, (size_t)B_ * MSLOT);
  k_zero<<<16,   256, 0, stream>>>(cbuf,  (size_t)B_ * H_);
  k_zero<<<16,   256, 0, stream>>>((float*)h16b, (size_t)B_ * H_ / 2);

  // ---- prep: bias sum, x conversion, weight packing ----
  k_addbias<<<4, 256, 0, stream>>>(bih, bhh, bsum);
  k_cvtx<<<16384, 256, 0, stream>>>(x, x16);
  k_packb<<<32, 256, 0, stream>>>(Wih, Wih_p, G4_, I_);        // 8192 threads
  k_packb<<<64, 256, 0, stream>>>(Whh, Whh_p, G4_, H_);        // 16384
  k_packb<<<16, 256, 0, stream>>>(Wp,  Wp_p,  H_,  H_);        // 4096
  k_packb<<<16, 256, 0, stream>>>(Wfc, Wfc_p, C_, 2 * H_);     // 4096

  // ---- hoisted input GEMM: gx[t][b][:] = x*W_ih^T + b_ih + b_hh ----
  // M=32768, N=1024, K=128 -> 131072 tiles / 8 waves = 16384 blocks
  k_gemm16<<<16384, 256, 0, stream>>>(x16, Wih_p, bsum, gx, T_ * B_, G4_, I_);

  // ---- sequential recurrence ----
  for (int t = 0; t < T_; ++t) {
    k_lstm_step<<<64, 128, 0, stream>>>(h16b, Whh_p, gx, cbuf, hl, hlstm16, t);
    k_attn_score<<<64, 32, 0, stream>>>(hl, s2, w_t, attnw, t);
    k_attn_apply<<<67, 256, 0, stream>>>(attnw, hl, memb, w_t, h16b, feat16,
                                         houts, viz, s2, t);
  }

  // ---- deferred big GEMMs ----
  // predicted_all = hlstm_all @ Wp^T + bp : M=32768, N=256, K=256 -> 32768 tiles
  k_gemm16<<<4096, 256, 0, stream>>>(hlstm16, Wp_p, bp, pred, B_ * T_, H_, H_);
  // out = [h_new | attn_c] @ Wfc^T + bfc : M=32768, N=128, K=512 -> 16384 tiles
  k_gemm16<<<2048, 256, 0, stream>>>(feat16, Wfc_p, bfc, out_fc, B_ * T_, C_, 2 * H_);
}
